// Net1_41695542509689
// MI455X (gfx1250) — matrix operand order repaired and checked by reading, not compiled, not run on hardware
//
#include <hip/hip_runtime.h>
#include <hip/hip_bf16.h>

typedef __attribute__((ext_vector_type(16))) __bf16 v16bf;
typedef __attribute__((ext_vector_type(8)))  float  v8f;
typedef __attribute__((ext_vector_type(4)))  float  v4f;

static constexpr int B_  = 32;
static constexpr int N_  = 2048;
static constexpr int F_  = 64;
static constexpr int C_  = 32;     // channels of both GCN layers
static constexpr int H_  = 512;
static constexpr int KSTEPS = N_ / 32;   // 64

// ---------------------------------------------------------------------------
// xw[b,n,:] = x[b,n,:] @ W   (W is [Fin,32]) ; no bias/relu here (applied later)
// ---------------------------------------------------------------------------
__global__ void dense_mm_kernel(const float* __restrict__ X,
                                const float* __restrict__ W,
                                float* __restrict__ Out, int Fin) {
    __shared__ float sW[F_ * C_];                     // up to 64*32 floats
    for (int i = threadIdx.x; i < Fin * C_; i += blockDim.x) sW[i] = W[i];
    __syncthreads();

    const int gid = blockIdx.x * blockDim.x + threadIdx.x;   // over B*N rows
    const float* xr = X + (size_t)gid * Fin;
    float acc[C_];
#pragma unroll
    for (int c = 0; c < C_; ++c) acc[c] = 0.0f;
    for (int f = 0; f < Fin; ++f) {
        const float xv = xr[f];
#pragma unroll
        for (int c = 0; c < C_; ++c) acc[c] = fmaf(xv, sW[f * C_ + c], acc[c]);
    }
    float* orow = Out + (size_t)gid * C_;
#pragma unroll
    for (int c = 0; c < C_; ++c) orow[c] = acc[c];
}

// ---------------------------------------------------------------------------
// Pack xw (f32 [B,N,32], N is the K dim of the big GEMM) into bf16 B-fragments.
// Layout: Bsw[b][ks][half][lane][elem], elem=0..15 (v16bf per lane).
//   K = ks*32 + (lane>>4)*8 + elem + (elem>=8 ? 8 : 0) ; Ncol = half*16 + (lane&15)
// (ISA 16-bit B 32x16: lanes 0-15 hold K {0..7,16..23}, lanes 16-31 hold K {8..15,24..31}.)
// ---------------------------------------------------------------------------
__global__ void pack_b_kernel(const float* __restrict__ XW,
                              __bf16* __restrict__ Bsw) {
    const int idx  = blockIdx.x * blockDim.x + threadIdx.x;  // B*KSTEPS*2*32
    const int lane = idx & 31;
    const int half = (idx >> 5) & 1;
    const int ks   = (idx >> 6) & (KSTEPS - 1);
    const int b    = idx >> 12;

    const int kbase = ks * 32 + ((lane >> 4) << 3);
    const int col   = half * 16 + (lane & 15);

    v16bf v;
#pragma unroll
    for (int e = 0; e < 16; ++e)
        v[e] = (__bf16)XW[(((size_t)b * N_) + (kbase + e + (e >= 8 ? 8 : 0))) * C_ + col];

    *(v16bf*)(Bsw + (size_t)idx * 16) = v;
}

// ---------------------------------------------------------------------------
// h[b,n,:] = relu( a[b,n,:] @ xw[b,:,:] + bias )  -- the hot kernel.
// One wave computes a 16(x32) output tile, accumulating over K in steps of 32
// with v_wmma_f32_16x16x32_bf16. A is streamed f32 -> bf16 in registers.
// ---------------------------------------------------------------------------
__global__ void __launch_bounds__(128)
gcn_wmma_kernel(const float* __restrict__ A,       // [B,N,N] f32
                const __bf16* __restrict__ Bsw,    // packed fragments
                const float* __restrict__ bias,    // [32]
                float* __restrict__ Out) {         // [B,N,32] f32
    const int lane = threadIdx.x & 31;
    const int wave = threadIdx.x >> 5;             // 0..3
    const int b    = blockIdx.y;
    const int n0   = (blockIdx.x * 4 + wave) * 16; // wave's 16-row tile

    // ISA 16-bit A 16x32 lane layout:
    //   lane<16 : row=lane,    holds K {0..7} and {16..23}
    //   lane>=16: row=lane-16, holds K {8..15} and {24..31}
    const int row  = n0 + (lane & 15);
    const int koff = (lane >> 4) << 3;             // 0 or 8

    const float*  arow  = A + ((size_t)b * N_ + row) * N_;
    const __bf16* bbase = Bsw + (size_t)b * KSTEPS * 2 * 32 * 16;

    v8f acc0 = {};
    v8f acc1 = {};

#pragma unroll 2
    for (int ks = 0; ks < KSTEPS; ++ks) {
        const int k0 = ks * 32 + koff;
        // 16 f32 of A per lane: [k0..k0+7] and [k0+16..k0+23], streamed (NT)
        const v4f a0 = __builtin_nontemporal_load((const v4f*)(arow + k0));
        const v4f a1 = __builtin_nontemporal_load((const v4f*)(arow + k0 + 4));
        const v4f a2 = __builtin_nontemporal_load((const v4f*)(arow + k0 + 16));
        const v4f a3 = __builtin_nontemporal_load((const v4f*)(arow + k0 + 20));

        v16bf af;
        af[0]  = (__bf16)a0[0]; af[1]  = (__bf16)a0[1];
        af[2]  = (__bf16)a0[2]; af[3]  = (__bf16)a0[3];
        af[4]  = (__bf16)a1[0]; af[5]  = (__bf16)a1[1];
        af[6]  = (__bf16)a1[2]; af[7]  = (__bf16)a1[3];
        af[8]  = (__bf16)a2[0]; af[9]  = (__bf16)a2[1];
        af[10] = (__bf16)a2[2]; af[11] = (__bf16)a2[3];
        af[12] = (__bf16)a3[0]; af[13] = (__bf16)a3[1];
        af[14] = (__bf16)a3[2]; af[15] = (__bf16)a3[3];

        const __bf16* bp = bbase + ((size_t)ks * 2 * 32 + lane) * 16;
        const v16bf bf0 = *(const v16bf*)(bp);             // cols 0..15
        const v16bf bf1 = *(const v16bf*)(bp + 32 * 16);   // cols 16..31

        acc0 = __builtin_amdgcn_wmma_f32_16x16x32_bf16(
                   false, af, false, bf0, (short)0, acc0, false, false);
        acc1 = __builtin_amdgcn_wmma_f32_16x16x32_bf16(
                   false, af, false, bf1, (short)0, acc1, false, false);
    }

    // C/D layout: VGPR r, lane l -> M = r + 8*(l>=16), Ncol = l&15
    const int mbase = n0 + ((lane >> 4) << 3);
    const int col   = lane & 15;
    const float bi0 = bias[col];
    const float bi1 = bias[col + 16];
#pragma unroll
    for (int r = 0; r < 8; ++r) {
        float v0 = acc0[r] + bi0; v0 = v0 > 0.0f ? v0 : 0.0f;
        float v1 = acc1[r] + bi1; v1 = v1 > 0.0f ? v1 : 0.0f;
        float* orow = Out + (((size_t)b * N_) + (mbase + r)) * C_;
        orow[col]      = v0;
        orow[col + 16] = v1;
    }
}

// ---------------------------------------------------------------------------
// pooled[b,c] = sum_n h[b,n,c]
// ---------------------------------------------------------------------------
__global__ void pool_kernel(const float* __restrict__ Hbuf,
                            float* __restrict__ pooled) {
    const int b = blockIdx.x;
    const int t = threadIdx.x;        // 256
    const int c = t & 31;
    const int seg = t >> 5;           // 0..7
    float s = 0.0f;
    for (int n = seg; n < N_; n += 8)
        s += Hbuf[((size_t)b * N_ + n) * C_ + c];
    __shared__ float red[256];
    red[t] = s;
    __syncthreads();
    if (seg == 0) {
        float tot = 0.0f;
#pragma unroll
        for (int i = 0; i < 8; ++i) tot += red[i * 32 + c];
        pooled[b * C_ + c] = tot;
    }
}

// ---------------------------------------------------------------------------
// out[b] = relu(pooled[b] @ Wf1 + bf1) @ Wf2 + bf2
// ---------------------------------------------------------------------------
__global__ void head_kernel(const float* __restrict__ pooled,
                            const float* __restrict__ Wf1,
                            const float* __restrict__ bf1,
                            const float* __restrict__ Wf2,
                            const float* __restrict__ bf2,
                            float* __restrict__ out) {
    const int b = blockIdx.x;
    const int t = threadIdx.x;        // 256
    __shared__ float sp[C_];
    if (t < C_) sp[t] = pooled[b * C_ + t];
    __syncthreads();
    float partial = 0.0f;
    for (int j = t; j < H_; j += 256) {
        float h = bf1[j];
#pragma unroll
        for (int c = 0; c < C_; ++c) h = fmaf(sp[c], Wf1[c * H_ + j], h);
        h = h > 0.0f ? h : 0.0f;
        partial += h * Wf2[j];
    }
    __shared__ float red[256];
    red[t] = partial;
    __syncthreads();
    for (int s = 128; s > 0; s >>= 1) {
        if (t < s) red[t] += red[t + s];
        __syncthreads();
    }
    if (t == 0) out[b] = red[0] + bf2[0];
}

// ---------------------------------------------------------------------------
extern "C" void kernel_launch(void* const* d_in, const int* in_sizes, int n_in,
                              void* d_out, int out_size, void* d_ws, size_t ws_size,
                              hipStream_t stream) {
    const float* x   = (const float*)d_in[0];
    const float* a   = (const float*)d_in[1];
    const float* W1  = (const float*)d_in[2];
    const float* b1  = (const float*)d_in[3];
    const float* W2  = (const float*)d_in[4];
    const float* b2  = (const float*)d_in[5];
    const float* Wf1 = (const float*)d_in[6];
    const float* bf1 = (const float*)d_in[7];
    const float* Wf2 = (const float*)d_in[8];
    const float* bf2 = (const float*)d_in[9];
    float* out = (float*)d_out;

    const size_t XW_BYTES = (size_t)B_ * N_ * C_ * sizeof(float);   // 8 MB
    char* ws = (char*)d_ws;
    float*  xw     = (float*)(ws);
    __bf16* bsw    = (__bf16*)(ws + XW_BYTES);
    float*  hbuf   = (float*)(ws + 2 * XW_BYTES);
    float*  pooled = (float*)(ws + 3 * XW_BYTES);

    const dim3 gcnGrid(N_ / 64, B_);     // 4 waves per wg, 16 rows each

    // ---- layer 1 ----
    dense_mm_kernel<<<(B_ * N_) / 256, 256, 0, stream>>>(x, W1, xw, F_);
    pack_b_kernel<<<(B_ * KSTEPS * 2 * 32) / 256, 256, 0, stream>>>(xw, bsw);
    gcn_wmma_kernel<<<gcnGrid, 128, 0, stream>>>(a, bsw, b1, hbuf);

    // ---- layer 2 ----
    dense_mm_kernel<<<(B_ * N_) / 256, 256, 0, stream>>>(hbuf, W2, xw, C_);
    pack_b_kernel<<<(B_ * KSTEPS * 2 * 32) / 256, 256, 0, stream>>>(xw, bsw);
    gcn_wmma_kernel<<<gcnGrid, 128, 0, stream>>>(a, bsw, b2, hbuf);

    // ---- pool + MLP head ----
    pool_kernel<<<B_, 256, 0, stream>>>(hbuf, pooled);
    head_kernel<<<B_, 256, 0, stream>>>(pooled, Wf1, bf1, Wf2, bf2, out);
}
